// VectorQuantizer_1597727834324
// MI455X (gfx1250) — matrix operand order repaired and checked by reading, not compiled, not run on hardware
//
#include <hip/hip_runtime.h>

#define NUM_EMB 512
#define EMB_DIM 64
#define BATCH   16
#define TLEN    8192
#define TILE_T  256
#define CBS     72   // padded halves per codebook row (144B, 16B aligned, odd dword stride/2 -> conflict-free)
#define XTS     72   // padded halves per x-tile row

typedef __attribute__((ext_vector_type(16))) _Float16     v16h;
typedef __attribute__((ext_vector_type(8)))  float        v8f;
typedef __attribute__((ext_vector_type(4)))  unsigned int u32x4;

union F16Frag { v16h h; u32x4 q[2]; };

__global__ __launch_bounds__(256, 1)
void vq_wmma_kernel(const float* __restrict__ x,
                    const float* __restrict__ cb,
                    float* __restrict__ out) {
  __shared__ alignas(16) _Float16 s_cb_hi[NUM_EMB * CBS];
  __shared__ alignas(16) _Float16 s_cb_lo[NUM_EMB * CBS];
  __shared__ alignas(16) _Float16 s_x_hi[TILE_T * XTS];
  __shared__ alignas(16) _Float16 s_x_lo[TILE_T * XTS];
  __shared__ float s_hn[NUM_EMB];
  __shared__ int   s_idx[TILE_T];

  const int tid = threadIdx.x;
  const int b   = blockIdx.x / (TLEN / TILE_T);
  const int t0  = (blockIdx.x % (TLEN / TILE_T)) * TILE_T;

  // ---- Phase 0: codebook -> f16 hi/lo split in LDS (padded), + exact f32 half-norms
  for (int i = tid; i < NUM_EMB * EMB_DIM; i += 256) {
    float c = cb[i];
    _Float16 hi = (_Float16)c;
    _Float16 lo = (_Float16)(c - (float)hi);
    int r = i >> 6, d = i & 63;
    s_cb_hi[r * CBS + d] = hi;
    s_cb_lo[r * CBS + d] = lo;
  }
  for (int r = tid; r < NUM_EMB; r += 256) {
    float s = 0.f;
    for (int d = 0; d < EMB_DIM; ++d) { float c = cb[r * EMB_DIM + d]; s += c * c; }
    s_hn[r] = 0.5f * s;
  }

  // ---- Phase 1: x tile -> transposed [t][d] hi/lo split in LDS (global reads coalesced along t)
  for (int d = 0; d < EMB_DIM; ++d) {
    float v = x[((size_t)b * EMB_DIM + d) * TLEN + (t0 + tid)];
    _Float16 hi = (_Float16)v;
    _Float16 lo = (_Float16)(v - (float)hi);
    s_x_hi[tid * XTS + d] = hi;
    s_x_lo[tid * XTS + d] = lo;
  }
  __syncthreads();

  const int lane  = tid & 31;
  const int wave  = tid >> 5;
  const int halfw = lane >> 4;   // 0: lanes 0-15, 1: lanes 16-31
  const int lcol  = lane & 15;

  // ---- Load all A fragments for this wave's two M-tiles (held in VGPRs)
  // A 16-bit 16x32 layout: lane<16 -> K chunks {0..7},{16..23}; lane>=16 -> {8..15},{24..31}
  F16Frag a_hi[2][2], a_lo[2][2];   // [mtile][kstep]
  for (int mt = 0; mt < 2; ++mt) {
    const int trow = (wave * 2 + mt) * 16 + lcol;
    for (int k = 0; k < 2; ++k) {
      const int k0 = (halfw ? 8 : 0) + 32 * k;
      const u32x4* ph = (const u32x4*)(s_x_hi + trow * XTS + k0);
      const u32x4* pl = (const u32x4*)(s_x_lo + trow * XTS + k0);
      a_hi[mt][k].q[0] = ph[0]; a_hi[mt][k].q[1] = ph[2];  // +16 halves
      a_lo[mt][k].q[0] = pl[0]; a_lo[mt][k].q[1] = pl[2];
    }
  }

  float bval[2][8];
  int   bidx[2][8];
  for (int mt = 0; mt < 2; ++mt)
    for (int v = 0; v < 8; ++v) { bval[mt][v] = -3.4e38f; bidx[mt][v] = 0; }

  // ---- Phase 2: sweep all 32 codebook N-tiles; 3-term split-f16 WMMA, f32 accumulate
  // B 16-bit 32x16 layout: lane<16 -> K chunks {0..7},{16..23}; lane>=16 -> {8..15},{24..31}
  for (int nt = 0; nt < 32; ++nt) {
    const int n = nt * 16 + lcol;
    F16Frag bh[2], bl[2];
    for (int k = 0; k < 2; ++k) {
      const int kb = (halfw ? 8 : 0) + 32 * k;
      const u32x4* ph = (const u32x4*)(s_cb_hi + n * CBS + kb);
      const u32x4* pl = (const u32x4*)(s_cb_lo + n * CBS + kb);
      bh[k].q[0] = ph[0]; bh[k].q[1] = ph[2];
      bl[k].q[0] = pl[0]; bl[k].q[1] = pl[2];
    }
    const float hn = s_hn[n];
    for (int mt = 0; mt < 2; ++mt) {
      v8f acc = {};
      for (int k = 0; k < 2; ++k) {
        acc = __builtin_amdgcn_wmma_f32_16x16x32_f16(false, a_hi[mt][k].h, false, bh[k].h,
                                                     (short)0, acc, false, false);
        acc = __builtin_amdgcn_wmma_f32_16x16x32_f16(false, a_lo[mt][k].h, false, bh[k].h,
                                                     (short)0, acc, false, false);
        acc = __builtin_amdgcn_wmma_f32_16x16x32_f16(false, a_hi[mt][k].h, false, bl[k].h,
                                                     (short)0, acc, false, false);
      }
      // C layout: lanes 0-15 VGPR v -> (M=v, N=lcol); lanes 16-31 -> (M=v+8, N=lcol)
      for (int v = 0; v < 8; ++v) {
        float s = acc[v] - hn;
        if (s > bval[mt][v]) { bval[mt][v] = s; bidx[mt][v] = n; }  // strict >: lowest idx wins ties
      }
    }
  }

  // ---- Phase 3: reduce argmax across the 16 lanes sharing each row (xor masks stay in half)
  for (int mt = 0; mt < 2; ++mt) {
    for (int v = 0; v < 8; ++v) {
      float bv = bval[mt][v];
      int   bi = bidx[mt][v];
      for (int m = 1; m <= 8; m <<= 1) {
        float ov = __shfl_xor(bv, m, 32);
        int   oi = __shfl_xor(bi, m, 32);
        if (ov > bv || (ov == bv && oi < bi)) { bv = ov; bi = oi; }
      }
      if (lcol == 0) s_idx[(wave * 2 + mt) * 16 + halfw * 8 + v] = bi;
    }
  }
  __syncthreads();

  // ---- Phase 4: gather + write both outputs, coalesced along t
  const int qi = s_idx[tid];
  const _Float16* rh = s_cb_hi + qi * CBS;
  const _Float16* rl = s_cb_lo + qi * CBS;
  const size_t obase   = (size_t)b * EMB_DIM * TLEN + (size_t)(t0 + tid);
  const size_t osecond = (size_t)BATCH * EMB_DIM * TLEN;
  for (int d = 0; d < EMB_DIM; ++d) {
    float v = (float)rh[d] + (float)rl[d];   // exact to ~5e-10 abs
    size_t o = obase + (size_t)d * TLEN;
    out[o]            = v;  // quantized_st (forward value == quantized)
    out[o + osecond]  = v;  // quantized
  }
}

extern "C" void kernel_launch(void* const* d_in, const int* in_sizes, int n_in,
                              void* d_out, int out_size, void* d_ws, size_t ws_size,
                              hipStream_t stream) {
  const float* x  = (const float*)d_in[0];
  const float* cb = (const float*)d_in[1];
  float* out = (float*)d_out;
  dim3 grid(BATCH * (TLEN / TILE_T));   // 512 blocks
  vq_wmma_kernel<<<grid, 256, 0, stream>>>(x, cb, out);
}
